// CrossAttention_71871982731569
// MI455X (gfx1250) — matrix operand order repaired and checked by reading, not compiled, not run on hardware
//
#include <hip/hip_runtime.h>
#include <hip/hip_bf16.h>

typedef _Float16 h16;
typedef __attribute__((ext_vector_type(16))) _Float16 v16h;
typedef __attribute__((ext_vector_type(8)))  _Float16 v8h;
typedef __attribute__((ext_vector_type(8)))  float    v8f;

#define D_MODEL 1024
#define NH 16
#define DH 64
#define TSEQ 2048
#define BATCH 2
#define MROWS (BATCH * TSEQ) /* 4096 */

// ---------------------------------------------------------------------------
// f32 -> f16 elementwise convert
// ---------------------------------------------------------------------------
__global__ void cvt_kernel(const float* __restrict__ in, h16* __restrict__ out, int n) {
    int i = blockIdx.x * blockDim.x + threadIdx.x;
    if (i < n) out[i] = (h16)in[i];
}

// Wt[n][k] = (f16) W[k][n]   (W is in x out, Wt is out x in so B-fragments are
// contiguous 32B row reads)
__global__ void wtrans_kernel(const float* __restrict__ W, h16* __restrict__ Wt) {
    int i = blockIdx.x * blockDim.x + threadIdx.x; // i = k*1024 + n
    int k = i >> 10, n = i & 1023;
    Wt[((size_t)n << 10) + k] = (h16)W[i];
}

// ---------------------------------------------------------------------------
// Tiled GEMM: out(M x 1024) = X(M x 1024, f16) @ Wt^T + bias
// One wave computes a 16(M) x 64(N) tile: A fragment reused across 4 WMMAs.
// MODE 0: f16 row-major output, scaled (Q: scale=1/8 folds in 1/sqrt(d_h))
// MODE 2: f16 per-head-transposed output Vt[b][h][d][enc]
// MODE 3: f32 row-major output (final projection -> d_out)
// ---------------------------------------------------------------------------
template <int MODE>
__global__ void gemm_kernel(const h16* __restrict__ X, const h16* __restrict__ Wt,
                            const float* __restrict__ bias, void* __restrict__ outp,
                            float scale) {
    const int lane  = threadIdx.x & 31;
    const int wid   = blockIdx.x * 4 + (threadIdx.x >> 5);
    const int mbase = (wid >> 4) * 16;       // D_MODEL/64 = 16 n-tiles per m row
    const int nbase = (wid & 15) * 64;
    const int nl    = lane & 15;
    const int hi    = lane >> 4;

    // A fragment (16x32 f16): lane = row m, halves = K kb..kb+15 (contiguous)
    const h16* arow  = X + ((size_t)(mbase + nl) << 10) + (hi ? 16 : 0);
    // B fragment (32x16 f16): lane = col n, halves = K kb2..kb2+15 (contiguous 32B)
    const h16* brow0 = Wt + ((size_t)(nbase + nl) << 10) + (hi ? 16 : 0);

    v8f c0 = {}, c1 = {}, c2 = {}, c3 = {};
    for (int ks = 0; ks < D_MODEL; ks += 32) {
        v8h alo = *(const v8h*)(arow + ks);
        v8h ahi = *(const v8h*)(arow + ks + 8);
        v16h a;
#pragma unroll
        for (int i = 0; i < 8; ++i) { a[i] = alo[i]; a[i + 8] = ahi[i]; }
        v16h b0 = *(const v16h*)(brow0 + ks);
        v16h b1 = *(const v16h*)(brow0 + (16 << 10) + ks);
        v16h b2 = *(const v16h*)(brow0 + (32 << 10) + ks);
        v16h b3 = *(const v16h*)(brow0 + (48 << 10) + ks);
        c0 = __builtin_amdgcn_wmma_f32_16x16x32_f16(false, a, false, b0, (short)0, c0, false, false);
        c1 = __builtin_amdgcn_wmma_f32_16x16x32_f16(false, a, false, b1, (short)0, c1, false, false);
        c2 = __builtin_amdgcn_wmma_f32_16x16x32_f16(false, a, false, b2, (short)0, c2, false, false);
        c3 = __builtin_amdgcn_wmma_f32_16x16x32_f16(false, a, false, b3, (short)0, c3, false, false);
    }

    v8f cc[4] = {c0, c1, c2, c3};
#pragma unroll
    for (int t = 0; t < 4; ++t) {
        int n = nbase + t * 16 + nl;       // C/D layout: lane holds column n
        float bv = bias[n];
#pragma unroll
        for (int r = 0; r < 8; ++r) {
            int m = mbase + r + 8 * hi;    // rows m = r + 8*(lane>=16)
            float val = cc[t][r] + bv;
            if constexpr (MODE == 0) {
                ((h16*)outp)[((size_t)m << 10) + n] = (h16)(val * scale);
            } else if constexpr (MODE == 2) {
                int bb = m >> 11, enc = m & 2047, hh = n >> 6, dd = n & 63;
                ((h16*)outp)[(((size_t)(bb * NH + hh) * DH + dd) << 11) + enc] = (h16)val;
            } else {
                ((float*)outp)[((size_t)m << 10) + n] = val;
            }
        }
    }
}

// ---------------------------------------------------------------------------
// Flash attention. One wave per (b, h, 16-query tile). Computes S^T tiles
// (A = K tile, B = Q^T fragment) so softmax reduces over the M index
// (8 regs + shfl_xor 16). Two consecutive S^T f32 C-tiles map lane-for-lane
// onto the f16 A-fragment layout of P for the P@V WMMA.
// ---------------------------------------------------------------------------
__global__ void attn_kernel(const h16* __restrict__ Qw, const h16* __restrict__ Kw,
                            const h16* __restrict__ Vt, h16* __restrict__ Yw) {
    const int lane = threadIdx.x & 31;
    const int nl   = lane & 15;
    const int hi   = lane >> 4;
    const int task  = blockIdx.x;        // B*H*(T_dec/16) tasks
    const int qtile = task & 127;
    const int h     = (task >> 7) & 15;
    const int b     = task >> 11;
    const int q0    = qtile << 4;

    // Q^T B-fragments (column n = query nl, K rows = head dim), contiguous 32B
    const h16* qrow = Qw + ((size_t)(b * TSEQ + q0 + nl) << 10) + h * DH + (hi ? 16 : 0);
    v16h bq0 = *(const v16h*)(qrow);        // d = 0..31
    v16h bq1 = *(const v16h*)(qrow + 32);   // d = 32..63

    const h16* kbase = Kw + ((size_t)(b * TSEQ) << 10) + h * DH + (hi ? 16 : 0);
    const h16* vbase = Vt + (((size_t)(b * NH + h) * DH + nl) << 11) + (hi ? 8 : 0);

    v8f acc0 = {}, acc1 = {}, acc2 = {}, acc3 = {};
    float mi = -3.0e38f, li = 0.0f;

    for (int eb = 0; eb < TSEQ; eb += 32) {
        v8f st[2];
#pragma unroll
        for (int sub = 0; sub < 2; ++sub) {
            const h16* krow = kbase + ((size_t)(eb + sub * 16 + nl) << 10);
            v8h l0 = *(const v8h*)(krow);
            v8h g0 = *(const v8h*)(krow + 8);
            v8h l1 = *(const v8h*)(krow + 32);
            v8h g1 = *(const v8h*)(krow + 40);
            v16h ak0, ak1;
#pragma unroll
            for (int i = 0; i < 8; ++i) {
                ak0[i] = l0[i]; ak0[i + 8] = g0[i];
                ak1[i] = l1[i]; ak1[i + 8] = g1[i];
            }
            v8f s = {};
            s = __builtin_amdgcn_wmma_f32_16x16x32_f16(false, ak0, false, bq0, (short)0, s, false, false);
            s = __builtin_amdgcn_wmma_f32_16x16x32_f16(false, ak1, false, bq1, (short)0, s, false, false);
            st[sub] = s;
        }

        // online softmax over enc (= M index of S^T)
        float mt = -3.0e38f;
#pragma unroll
        for (int r = 0; r < 8; ++r) mt = fmaxf(mt, fmaxf(st[0][r], st[1][r]));
        mt = fmaxf(mt, __shfl_xor(mt, 16, 32));
        float mnew = fmaxf(mi, mt);
        float corr = __expf(mi - mnew);
        float psum = 0.0f;
        v16h ap;   // P A-fragment: halves 0..7 <- st[0], 8..15 <- st[1] (same lane)
#pragma unroll
        for (int r = 0; r < 8; ++r) {
            float p0 = __expf(st[0][r] - mnew);
            float p1 = __expf(st[1][r] - mnew);
            psum += p0 + p1;
            ap[r]     = (h16)p0;
            ap[r + 8] = (h16)p1;
        }
        psum += __shfl_xor(psum, 16, 32);
        li = li * corr + psum;
        mi = mnew;

        // rescale accumulators: row r of acc corresponds to query r + 8*hi,
        // whose corr lives in lane (r + 8*hi)
#pragma unroll
        for (int r = 0; r < 8; ++r) {
            float cr = __shfl(corr, r + 8 * hi, 32);
            acc0[r] *= cr; acc1[r] *= cr; acc2[r] *= cr; acc3[r] *= cr;
        }

        // V B-fragments from per-head-transposed layout (halves 0..7 = K 8h.., 8..15 = K 16+8h..)
        v16h bvs[4];
#pragma unroll
        for (int j = 0; j < 4; ++j) {
            const h16* vp = vbase + ((size_t)(16 * j) << 11) + eb;
            v8h vl = *(const v8h*)(vp);
            v8h vh = *(const v8h*)(vp + 16);
#pragma unroll
            for (int i = 0; i < 8; ++i) { bvs[j][i] = vl[i]; bvs[j][i + 8] = vh[i]; }
        }
        v16h bv0 = bvs[0], bv1 = bvs[1], bv2 = bvs[2], bv3 = bvs[3];
        acc0 = __builtin_amdgcn_wmma_f32_16x16x32_f16(false, ap, false, bv0, (short)0, acc0, false, false);
        acc1 = __builtin_amdgcn_wmma_f32_16x16x32_f16(false, ap, false, bv1, (short)0, acc1, false, false);
        acc2 = __builtin_amdgcn_wmma_f32_16x16x32_f16(false, ap, false, bv2, (short)0, acc2, false, false);
        acc3 = __builtin_amdgcn_wmma_f32_16x16x32_f16(false, ap, false, bv3, (short)0, acc3, false, false);
    }

    float linv = 1.0f / li;
    h16* yrow = Yw + ((size_t)(b * TSEQ + q0) << 10) + h * DH + nl;
#pragma unroll
    for (int r = 0; r < 8; ++r) {
        float lr = __shfl(linv, r + 8 * hi, 32);
        size_t row = (size_t)(r + 8 * hi) << 10;
        yrow[row + 0]  = (h16)(acc0[r] * lr);
        yrow[row + 16] = (h16)(acc1[r] * lr);
        yrow[row + 32] = (h16)(acc2[r] * lr);
        yrow[row + 48] = (h16)(acc3[r] * lr);
    }
}

// ---------------------------------------------------------------------------
extern "C" void kernel_launch(void* const* d_in, const int* in_sizes, int n_in,
                              void* d_out, int out_size, void* d_ws, size_t ws_size,
                              hipStream_t stream) {
    const float* tgt = (const float*)d_in[0];
    const float* mem = (const float*)d_in[1];
    const float* W_q = (const float*)d_in[2];
    const float* b_q = (const float*)d_in[3];
    const float* W_k = (const float*)d_in[4];
    const float* b_k = (const float*)d_in[5];
    const float* W_v = (const float*)d_in[6];
    const float* b_v = (const float*)d_in[7];
    const float* W_o = (const float*)d_in[8];
    const float* b_o = (const float*)d_in[9];

    char* ws = (char*)d_ws;
    const size_t MB = 1024 * 1024;
    h16* Xdec = (h16*)(ws + 0 * MB);   // 8 MB (reused as Yw after Q proj)
    h16* Xenc = (h16*)(ws + 8 * MB);   // 8 MB
    h16* Qw   = (h16*)(ws + 16 * MB);  // 8 MB
    h16* Kw   = (h16*)(ws + 24 * MB);  // 8 MB
    h16* Vt   = (h16*)(ws + 32 * MB);  // 8 MB
    h16* Wtq  = (h16*)(ws + 40 * MB);  // 2 MB each
    h16* Wtk  = (h16*)(ws + 42 * MB);
    h16* Wtv  = (h16*)(ws + 44 * MB);
    h16* Wto  = (h16*)(ws + 46 * MB);  // total 48 MB
    h16* Yw   = Xdec;                  // alias: Xdec dead after Q projection

    const int nact = MROWS * D_MODEL;          // 4,194,304
    cvt_kernel<<<nact / 256, 256, 0, stream>>>(tgt, Xdec, nact);
    cvt_kernel<<<nact / 256, 256, 0, stream>>>(mem, Xenc, nact);
    const int nw = D_MODEL * D_MODEL;          // 1,048,576
    wtrans_kernel<<<nw / 256, 256, 0, stream>>>(W_q, Wtq);
    wtrans_kernel<<<nw / 256, 256, 0, stream>>>(W_k, Wtk);
    wtrans_kernel<<<nw / 256, 256, 0, stream>>>(W_v, Wtv);
    wtrans_kernel<<<nw / 256, 256, 0, stream>>>(W_o, Wto);

    const int gemm_blocks = (MROWS / 16) * (D_MODEL / 64) / 4; // 1024 blocks x 4 waves
    gemm_kernel<0><<<gemm_blocks, 128, 0, stream>>>(Xdec, Wtq, b_q, Qw, 0.125f); // 1/sqrt(64)
    gemm_kernel<0><<<gemm_blocks, 128, 0, stream>>>(Xenc, Wtk, b_k, Kw, 1.0f);
    gemm_kernel<2><<<gemm_blocks, 128, 0, stream>>>(Xenc, Wtv, b_v, Vt, 1.0f);

    attn_kernel<<<BATCH * NH * (TSEQ / 16), 32, 0, stream>>>(Qw, Kw, Vt, Yw);

    gemm_kernel<3><<<gemm_blocks, 128, 0, stream>>>(Yw, Wto, b_o, d_out, 1.0f);
}
